// QRN_71597104824739
// MI455X (gfx1250) — compile-verified, matrix operand order checked by reading, not run on hardware
//
#include <hip/hip_runtime.h>
#include <hip/hip_bf16.h>

// ---------------- problem constants (from reference) ----------------
#define T_LEN   65536
#define DIM     256
#define DIM2    512
#define LAYERS  3
#define BLK_M   64          // rows of T per block in gates kernel
#define KC      32          // K-chunk per WMMA stage
#define NCHUNK  (DIM / KC)  // 8 K-chunks for a 256-deep GEMM
#define WSLICE  (DIM * KC)  // 8192 bf16 = 16KB, one weight K-slice
#define NSEG    512         // scan segments
#define SEGLEN  128         // steps per segment (NSEG*SEGLEN == T_LEN)

typedef __attribute__((ext_vector_type(16))) __bf16        v16bf;
typedef __attribute__((ext_vector_type(8)))  float         v8f;
typedef __attribute__((ext_vector_type(4)))  unsigned int  v4u;

__device__ __forceinline__ unsigned short f2bf(float f) {
    unsigned int x = __float_as_uint(f);
    x += 0x7FFFu + ((x >> 16) & 1u);      // round-to-nearest-even
    return (unsigned short)(x >> 16);
}
__device__ __forceinline__ float sigmoid_f(float x) {
    return 1.0f / (1.0f + __expf(-x));
}
__device__ __forceinline__ float tanh_f(float x) {
    float e = __expf(2.0f * x);           // 1 - 2/(e^{2x}+1): stable at +/- inf
    return 1.0f - 2.0f / (e + 1.0f);
}

// ---- CDNA5 async global->LDS copy (ASYNCcnt), inline asm per bridge doc ----
__device__ __forceinline__ void async_b128(unsigned lds_off, const void* g) {
    asm volatile("global_load_async_to_lds_b128 %0, %1, off"
                 :: "v"(lds_off), "v"((unsigned long long)(uintptr_t)g)
                 : "memory");
}
__device__ __forceinline__ void wait_async_0()  { asm volatile("s_wait_asynccnt 0x0" ::: "memory"); }
__device__ __forceinline__ void wait_async_12() { asm volatile("s_wait_asynccnt 0xc" ::: "memory"); }

// Gather a v16bf WMMA fragment from LDS as two aligned 16B runs
// (dwords 4h..4h+3 and 8+4h..8+4h+3) -> exactly 2x ds_load_b128.
// A: [row][DIM] row-major bf16 (stride=DIM, kc=chunk base)
// B: [n][KC] K-transposed slice (stride=KC, kc=0)
__device__ __forceinline__ v16bf load_frag(const unsigned short* __restrict__ base,
                                           int row, int stride, int kc, int half) {
    union { v4u q[2]; v16bf v; } f;
    const v4u* p = (const v4u*)(base + row * stride + kc);   // 16B-aligned
    f.q[0] = p[half];
    f.q[1] = p[2 + half];
    return f.v;
}

// Copy one pre-swizzled 16KB weight K-slice global->LDS (4 x b128 per thread).
__device__ __forceinline__ void stage_w_slice(unsigned short* ldsDst,
                                              const unsigned short* g, int tid) {
    unsigned lbase = (unsigned)(uintptr_t)ldsDst;
    const char* gb = (const char*)g;
#pragma unroll
    for (int j = 0; j < 4; ++j) {
        unsigned off = (unsigned)(tid + 256 * j) * 16u;
        async_b128(lbase + off, gb + off);
    }
}

// Copy a 64-row x 256-col bf16 A-panel global->LDS with direction mapping.
__device__ __forceinline__ void stage_a_panel(unsigned short* ldsDst,
                                              const unsigned short* g,
                                              int s_base, int dir, int tid) {
    unsigned lbase = (unsigned)(uintptr_t)ldsDst;
#pragma unroll
    for (int j = 0; j < 8; ++j) {
        int c    = tid + 256 * j;          // 0..2047 16B-chunks
        int row  = c >> 5;                 // 0..63
        int coff = (c & 31) * 16;          // byte offset within 512B row
        int s = s_base + row;
        int t = dir ? (T_LEN - 1 - s) : s;
        async_b128(lbase + (unsigned)(row * 512 + coff),
                   (const char*)g + (size_t)t * 512 + coff);
    }
}

// ---------------- one-time / per-layer bf16 prep ----------------
__global__ void __launch_bounds__(256)
qrn_prep_x(const float* __restrict__ x, unsigned short* __restrict__ x_bf, int n) {
    int i = blockIdx.x * 256 + threadIdx.x;
    if (i < n) x_bf[i] = f2bf(x[i]);
}

__global__ void __launch_bounds__(256)
qrn_prep_layer(const float* __restrict__ x, const float* __restrict__ qbuf,
               const float* __restrict__ qvec, int layer0,
               unsigned short* __restrict__ q_bf, unsigned short* __restrict__ u_bf) {
    int i = blockIdx.x * 256 + threadIdx.x;
    float q = layer0 ? qvec[i & (DIM - 1)] : qbuf[i];
    q_bf[i] = f2bf(q);
    u_bf[i] = f2bf(x[i] * q);
}

// Swizzle weights [K][N] f32 -> per-K-chunk LDS image bf16: [ko][n][kr].
__global__ void __launch_bounds__(256)
qrn_swizzle_w(const float* __restrict__ in, unsigned short* __restrict__ out,
              int kdim, int n_total) {
    int o = blockIdx.x * 256 + threadIdx.x;
    if (o >= n_total) return;
    const int per_layer = kdim * DIM;
    const int layer = o / per_layer;
    const int r = o - layer * per_layer;
    const int kr = r & (KC - 1);
    const int n  = (r >> 5) & (DIM - 1);
    const int ko = r >> 13;                      // r / (KC*DIM)
    out[o] = f2bf(in[(size_t)layer * per_layer + (size_t)(ko * KC + kr) * DIM + n]);
}

// ---------------- fused gates kernel (WMMA bf16, async-LDS pipelined) ----------
// For 64 scan-order rows s (t = dir ? T-1-s : s):
//   zpre = u @ Wz ; hpre = x @ Wh[0:256] + q @ Wh[256:512]
//   a_out[s] = 1 - sigmoid(zpre+bz) ; b_out[s] = sigmoid(zpre+bz)*tanh(hpre+bh)
__global__ void __launch_bounds__(256)
qrn_gates(const unsigned short* __restrict__ x_bf,   // [T,D] bf16 time order
          const unsigned short* __restrict__ q_bf,   // [T,D] bf16 time order
          const unsigned short* __restrict__ u_bf,   // [T,D] bf16 time order
          const unsigned short* __restrict__ Wzs,    // [8][256][32] swizzled bf16
          const unsigned short* __restrict__ Whs,    // [16][256][32] swizzled bf16
          const float* __restrict__ bz,
          const float* __restrict__ bh,
          float* __restrict__ a_out,                 // [T,D] scan order
          float* __restrict__ b_out,                 // [T,D] scan order
          int dir)
{
    extern __shared__ __align__(16) unsigned short smem[];   // 192KB dynamic LDS
    unsigned short* ldsX = smem;                     // 64*256 bf16 (32KB)
    unsigned short* ldsQ = ldsX + BLK_M * DIM;
    unsigned short* ldsU = ldsQ + BLK_M * DIM;
    unsigned short* ldsW = ldsU + BLK_M * DIM;       // 2 bufs * 3 * WSLICE (96KB)

    const int tid  = threadIdx.x;
    const int lane = tid & 31;
    const int w    = tid >> 5;       // 8 waves
    const int wm   = w & 3;          // 4 waves in M (16 rows each)
    const int wn   = w >> 2;         // 2 waves in N (128 cols each)
    const int s_base = blockIdx.x * BLK_M;

    // ---- pipeline prologue: A panel (24 asyncs) + weight slice 0 (12 asyncs) ----
    stage_a_panel(ldsX, x_bf, s_base, dir, tid);
    stage_a_panel(ldsQ, q_bf, s_base, dir, tid);
    stage_a_panel(ldsU, u_bf, s_base, dir, tid);
    stage_w_slice(ldsW + 0 * WSLICE, Wzs,              tid);
    stage_w_slice(ldsW + 1 * WSLICE, Whs,              tid);
    stage_w_slice(ldsW + 2 * WSLICE, Whs + 8 * WSLICE, tid);

    v8f zacc[8] = {};
    v8f hacc[8] = {};
    const int arow  = wm * 16 + (lane & 15);
    const int half  = lane >> 4;
    const int bcol0 = wn * 128 + (lane & 15);

    for (int ko = 0; ko < NCHUNK; ++ko) {
        const int cur = ko & 1;
        if (ko + 1 < NCHUNK) {
            // issue next slice into the other buffer, then wait for current (<=12 left)
            unsigned short* wb = ldsW + (cur ^ 1) * 3 * WSLICE;
            stage_w_slice(wb + 0 * WSLICE, Wzs + (ko + 1) * WSLICE,       tid);
            stage_w_slice(wb + 1 * WSLICE, Whs + (ko + 1) * WSLICE,       tid);
            stage_w_slice(wb + 2 * WSLICE, Whs + (8 + ko + 1) * WSLICE,   tid);
            wait_async_12();
        } else {
            wait_async_0();
        }
        __syncthreads();

        const unsigned short* wbuf = ldsW + cur * 3 * WSLICE;
        const int kc = ko * KC;
        const v16bf fu = load_frag(ldsU, arow, DIM, kc, half);
        const v16bf fx = load_frag(ldsX, arow, DIM, kc, half);
        const v16bf fq = load_frag(ldsQ, arow, DIM, kc, half);

        // software-pipelined B fragments: next tile's LDS loads fill the
        // WMMA->VALU WAR hazard shadow (no v_nop padding needed)
        v16bf bwz = load_frag(wbuf + 0 * WSLICE, bcol0, KC, 0, half);
        v16bf bwt = load_frag(wbuf + 1 * WSLICE, bcol0, KC, 0, half);
        v16bf bwb = load_frag(wbuf + 2 * WSLICE, bcol0, KC, 0, half);
#pragma unroll
        for (int nt = 0; nt < 8; ++nt) {
            v16bf nz = bwz, nt_ = bwt, nb = bwb;
            if (nt + 1 < 8) {
                const int bcol = bcol0 + (nt + 1) * 16;
                bwz = load_frag(wbuf + 0 * WSLICE, bcol, KC, 0, half);
                bwt = load_frag(wbuf + 1 * WSLICE, bcol, KC, 0, half);
                bwb = load_frag(wbuf + 2 * WSLICE, bcol, KC, 0, half);
            }
            zacc[nt] = __builtin_amdgcn_wmma_f32_16x16x32_bf16(
                false, fu, false, nz, (short)0, zacc[nt], false, false);
            hacc[nt] = __builtin_amdgcn_wmma_f32_16x16x32_bf16(
                false, fx, false, nt_, (short)0, hacc[nt], false, false);
            hacc[nt] = __builtin_amdgcn_wmma_f32_16x16x32_bf16(
                false, fq, false, nb, (short)0, hacc[nt], false, false);
        }
        __syncthreads();
    }

    // ---- epilogue: bias + sigmoid/tanh + a,b stores ----
#pragma unroll
    for (int nt = 0; nt < 8; ++nt) {
        const int n = wn * 128 + nt * 16 + (lane & 15);
        const float bzv = bz[n];
        const float bhv = bh[n];
#pragma unroll
        for (int v = 0; v < 8; ++v) {
            const int m = wm * 16 + (lane >> 4) * 8 + v;   // C-layout: VGPR v, lane half
            const int s = s_base + m;
            const float z  = sigmoid_f(zacc[nt][v] + bzv);
            const float ht = tanh_f(hacc[nt][v] + bhv);
            a_out[s * DIM + n] = 1.0f - z;
            b_out[s * DIM + n] = z * ht;
        }
    }
}

// ---------------- segmented linear-recurrence scan ----------------
__global__ void __launch_bounds__(256)
qrn_scan_local(const float* __restrict__ a, const float* __restrict__ b,
               float* __restrict__ segA, float* __restrict__ segB)
{
    const int d   = threadIdx.x;
    const int seg = blockIdx.x;
    float A = 1.0f, B = 0.0f;
    int s = seg * SEGLEN;
    for (int i = 0; i < SEGLEN; ++i, ++s) {
        const float av = a[s * DIM + d];
        const float bv = b[s * DIM + d];
        A = av * A;
        B = av * B + bv;
    }
    segA[seg * DIM + d] = A;
    segB[seg * DIM + d] = B;
}

__global__ void __launch_bounds__(256)
qrn_scan_carry(const float* __restrict__ segA, const float* __restrict__ segB,
               float* __restrict__ carry)
{
    const int d = threadIdx.x;
    float C = 0.0f;
    for (int seg = 0; seg < NSEG; ++seg) {
        carry[seg * DIM + d] = C;
        C = segA[seg * DIM + d] * C + segB[seg * DIM + d];
    }
}

__global__ void __launch_bounds__(256)
qrn_scan_fix(const float* __restrict__ a, float* __restrict__ b_h,
             const float* __restrict__ carry)
{
    const int d   = threadIdx.x;
    const int seg = blockIdx.x;
    float h = carry[seg * DIM + d];
    int s = seg * SEGLEN;
    for (int i = 0; i < SEGLEN; ++i, ++s) {
        h = a[s * DIM + d] * h + b_h[s * DIM + d];
        b_h[s * DIM + d] = h;
    }
}

// ---------------- q_{t}^{k+1} = h_fwd[t] + h_bwd_scanorder[T-1-t] -------------
__global__ void __launch_bounds__(256)
qrn_combine(const float* __restrict__ hf, const float* __restrict__ hb,
            float* __restrict__ qnext)
{
    const int idx = blockIdx.x * 256 + threadIdx.x;
    const int t = idx / DIM;
    const int d = idx - t * DIM;
    qnext[idx] = hf[idx] + hb[(T_LEN - 1 - t) * DIM + d];
}

__global__ void __launch_bounds__(256)
qrn_final(const float* __restrict__ hf, float* __restrict__ out)
{
    const int d = threadIdx.x;
    out[d] = hf[(T_LEN - 1) * DIM + d];
}

// ---------------- host orchestration ----------------
extern "C" void kernel_launch(void* const* d_in, const int* in_sizes, int n_in,
                              void* d_out, int out_size, void* d_ws, size_t ws_size,
                              hipStream_t stream)
{
    (void)in_sizes; (void)n_in; (void)out_size; (void)ws_size;

    const float* story    = (const float*)d_in[0];   // [T,D]
    const float* question = (const float*)d_in[1];   // [D]
    const float* Wz_f = (const float*)d_in[2];       // [L,D,D]
    const float* bz_f = (const float*)d_in[3];       // [L,D]
    const float* Wh_f = (const float*)d_in[4];       // [L,2D,D]
    const float* bh_f = (const float*)d_in[5];       // [L,D]
    const float* Wz_b = (const float*)d_in[6];
    const float* bz_b = (const float*)d_in[7];
    const float* Wh_b = (const float*)d_in[8];
    const float* bh_b = (const float*)d_in[9];

    const size_t TD = (size_t)T_LEN * DIM;
    float* qbuf = (float*)d_ws;          // current-layer q  [T,D]
    float* hf   = qbuf + TD;             // fwd gates-b / h_fwd
    float* ab   = hf + TD;               // gates a
    float* hb   = ab + TD;               // bwd gates-b / h_bwd
    float* segA  = hb + TD;
    float* segB  = segA + (size_t)NSEG * DIM;
    float* carry = segB + (size_t)NSEG * DIM;
    unsigned short* wzf = (unsigned short*)(carry + (size_t)NSEG * DIM);
    unsigned short* whf = wzf + (size_t)LAYERS * DIM * DIM;
    unsigned short* wzb = whf + (size_t)LAYERS * DIM2 * DIM;
    unsigned short* whb = wzb + (size_t)LAYERS * DIM * DIM;
    unsigned short* x_bf = whb + (size_t)LAYERS * DIM2 * DIM;
    unsigned short* q_bf = x_bf + TD;
    unsigned short* u_bf = q_bf + TD;

    const int nWz = LAYERS * DIM * DIM;
    const int nWh = LAYERS * DIM2 * DIM;
    qrn_swizzle_w<<<(nWz + 255) / 256, 256, 0, stream>>>(Wz_f, wzf, DIM,  nWz);
    qrn_swizzle_w<<<(nWh + 255) / 256, 256, 0, stream>>>(Wh_f, whf, DIM2, nWh);
    qrn_swizzle_w<<<(nWz + 255) / 256, 256, 0, stream>>>(Wz_b, wzb, DIM,  nWz);
    qrn_swizzle_w<<<(nWh + 255) / 256, 256, 0, stream>>>(Wh_b, whb, DIM2, nWh);
    qrn_prep_x<<<(int)(TD / 256), 256, 0, stream>>>(story, x_bf, (int)TD);

    const int gatesGrid = T_LEN / BLK_M;                 // 1024 blocks
    const int combGrid  = (int)(TD / 256);
    const unsigned smemBytes = (3 * BLK_M * DIM + 2 * 3 * WSLICE) * sizeof(unsigned short); // 192KB

    for (int l = 0; l < LAYERS; ++l) {
        const unsigned short* wz_fl = wzf + (size_t)l * DIM * DIM;
        const unsigned short* wh_fl = whf + (size_t)l * DIM2 * DIM;
        const unsigned short* wz_bl = wzb + (size_t)l * DIM * DIM;
        const unsigned short* wh_bl = whb + (size_t)l * DIM2 * DIM;
        const int layer0 = (l == 0) ? 1 : 0;

        // q / u bf16 prep (shared by both directions)
        qrn_prep_layer<<<combGrid, 256, 0, stream>>>(story, qbuf, question, layer0,
                                                     q_bf, u_bf);

        // forward direction -> h_fwd in hf
        qrn_gates<<<gatesGrid, 256, smemBytes, stream>>>(x_bf, q_bf, u_bf,
            wz_fl, wh_fl, bz_f + l * DIM, bh_f + l * DIM, ab, hf, /*dir*/0);
        qrn_scan_local<<<NSEG, 256, 0, stream>>>(ab, hf, segA, segB);
        qrn_scan_carry<<<1, 256, 0, stream>>>(segA, segB, carry);
        qrn_scan_fix<<<NSEG, 256, 0, stream>>>(ab, hf, carry);

        // backward direction -> h_bwd (scan order) in hb
        qrn_gates<<<gatesGrid, 256, smemBytes, stream>>>(x_bf, q_bf, u_bf,
            wz_bl, wh_bl, bz_b + l * DIM, bh_b + l * DIM, ab, hb, /*dir*/1);
        qrn_scan_local<<<NSEG, 256, 0, stream>>>(ab, hb, segA, segB);
        qrn_scan_carry<<<1, 256, 0, stream>>>(segA, segB, carry);
        qrn_scan_fix<<<NSEG, 256, 0, stream>>>(ab, hb, carry);

        if (l < LAYERS - 1)
            qrn_combine<<<combGrid, 256, 0, stream>>>(hf, hb, qbuf);
    }

    qrn_final<<<1, 256, 0, stream>>>(hf, (float*)d_out);
}
